// LlamaAttention_40132174414037
// MI455X (gfx1250) — matrix operand order repaired and checked by reading, not compiled, not run on hardware
//
#include <hip/hip_runtime.h>

typedef __attribute__((ext_vector_type(16))) __bf16 v16bf;
typedef __attribute__((ext_vector_type(8)))  __bf16 v8bf;
typedef __attribute__((ext_vector_type(8)))  float  v8f;
typedef __attribute__((ext_vector_type(4)))  unsigned int v4u;
typedef __attribute__((ext_vector_type(8)))  int    v8i;
typedef __attribute__((ext_vector_type(4)))  int    v4i;

#define HIDDEN 4096
#define NHEADS 32
#define NKV    8
#define HD     128
#define BATCH  2
#define SEQ    2048
#define BS     (BATCH*SEQ)      /* 4096 */
#define KVDIM  (NKV*HD)         /* 1024 */
#define GQA    (NHEADS/NKV)     /* 4 */

// ---------------------------------------------------------------- WMMA helper
__device__ __forceinline__ v8f wmma_bf16(v16bf a, v16bf b, v8f c) {
  // D = A(16x32 bf16) * B(32x16 bf16) + C(16x16 f32)
  return __builtin_amdgcn_wmma_f32_16x16x32_bf16(false, a, false, b,
                                                 (short)0, c, false, false);
}

// ------------------------------------------------- gfx1250 async global->LDS
__device__ __forceinline__ void async_copy_b128(unsigned lds_off, const void* gptr) {
  asm volatile("global_load_async_to_lds_b128 %0, %1, off"
               :: "v"(lds_off), "v"(gptr) : "memory");
}
__device__ __forceinline__ void wait_async0() {
  asm volatile("s_wait_asynccnt 0x0" ::: "memory");
}
__device__ __forceinline__ void wait_async2() {
  asm volatile("s_wait_asynccnt 0x2" ::: "memory");
}

// ------------------------------------------------- Tensor Data Mover (2D tile)
// Loads an (nrows x tileK) bf16 tile starting at element (row0, col0) of a
// row-major (tensorN x tensorK) matrix into LDS at byte offset lds_off.
__device__ __forceinline__ void tdm_load_tile(const __bf16* base, long row0, long col0,
                                              int nrows, int tensorN, int tensorK,
                                              int tileK, unsigned lds_off)
{
  const unsigned long long ga =
      (unsigned long long)(const void*)(base + row0 * (long)tensorK + col0);
  v4u g0;
  g0.x = 1u;                                            // count=1, load, user mode
  g0.y = lds_off;                                       // LDS byte address
  g0.z = (unsigned)(ga & 0xFFFFFFFFu);                  // global_addr[31:0]
  g0.w = (unsigned)((ga >> 32) & 0x01FFFFFFu)           // global_addr[56:32]
       | (2u << 30);                                    // type = 2 ("image")
  v8i g1;
  g1[0] = (1 << 16);                                    // data_size=2B, wg_mask=0
  g1[1] = (int)(((unsigned)tensorK & 0xFFFFu) << 16);   // tensor_dim0[15:0]
  g1[2] = (int)((((unsigned)tensorK >> 16) & 0xFFFFu)   // tensor_dim0[31:16]
       | (((unsigned)tensorN & 0xFFFFu) << 16));        // tensor_dim1[15:0]
  g1[3] = (int)((((unsigned)tensorN >> 16) & 0xFFFFu)   // tensor_dim1[31:16]
       | ((unsigned)tileK << 16));                      // tile_dim0 = K-slab
  g1[4] = nrows;                                        // tile_dim1; tile_dim2=0
  g1[5] = tensorK;                                      // tensor_dim0_stride[31:0]
  g1[6] = 0;                                            // stride hi / dim1_stride
  g1[7] = 0;
  const v4i z4 = (v4i)0;                                // 2D: groups 2/3 unused
  const v8i z8 = (v8i)0;
  // clang-23 / therock-10.0 form: 6 args (g0, g1, g2, g3, extra, cpol)
  __builtin_amdgcn_tensor_load_to_lds(g0, g1, z4, z4, z8, 0);
}

// --------------------------------------- 16-lane group reductions (C-layout)
__device__ __forceinline__ float redmax16(float v) {
  v = fmaxf(v, __shfl_xor(v, 1, 16));
  v = fmaxf(v, __shfl_xor(v, 2, 16));
  v = fmaxf(v, __shfl_xor(v, 4, 16));
  v = fmaxf(v, __shfl_xor(v, 8, 16));
  return v;
}
__device__ __forceinline__ float redsum16(float v) {
  v += __shfl_xor(v, 1, 16);
  v += __shfl_xor(v, 2, 16);
  v += __shfl_xor(v, 4, 16);
  v += __shfl_xor(v, 8, 16);
  return v;
}

// ============================================================================
// GEMM: C[M,N] = A[M,K] * B[N,K]^T   (A,B bf16 row-major, C f32)
// Block 256 thr = 8 waves; block tile 128x256, K-step 32, double-buffered LDS.
// Wave tile 64x64 => 16 WMMAs per K-step.  A staged via per-lane async copies,
// B staged via the Tensor Data Mover (wave 0 issues one D# per K-step).
// ============================================================================
#define BM  128
#define BN2 256
#define BK  32

__global__ __launch_bounds__(256) void gemm_abt_bf16(
    const __bf16* __restrict__ A, const __bf16* __restrict__ Bw,
    float* __restrict__ C, int M, int N, int K)
{
  __shared__ __align__(64) __bf16 As[2][BM * BK];    // 2 x 8 KB
  __shared__ __align__(64) __bf16 Bs[2][BN2 * BK];   // 2 x 16 KB
  const int tid  = threadIdx.x;
  const int lane = tid & 31;
  const int wave = tid >> 5;
  const int ln   = lane & 15;   // fragment column / row-low selector
  const int hh   = lane >> 4;   // lane-half selector
  const int wm   = wave & 1;    // 2 wave rows * 64 = 128
  const int wn   = wave >> 1;   // 4 wave cols * 64 = 256
  const long bm0 = (long)blockIdx.y * BM;
  const long bn0 = (long)blockIdx.x * BN2;

  v8f acc[4][4];
#pragma unroll
  for (int i = 0; i < 4; ++i)
#pragma unroll
    for (int j = 0; j < 4; ++j) acc[i][j] = (v8f)(0.0f);

  const unsigned asAo[2] = { (unsigned)(unsigned long long)(void*)&As[0][0],
                             (unsigned)(unsigned long long)(void*)&As[1][0] };
  const unsigned asBo[2] = { (unsigned)(unsigned long long)(void*)&Bs[0][0],
                             (unsigned)(unsigned long long)(void*)&Bs[1][0] };

  // Stage one K-slab into LDS buffer `buf`: A via async copies, B via TDM.
  auto stage = [&](int buf, int kt) {
#pragma unroll
    for (int r = 0; r < 2; ++r) {           // 128x32 bf16 = 512 x 16B chunks
      const int c = tid + (r << 8);
      async_copy_b128(asAo[buf] + (unsigned)c * 16u,
                      A + (bm0 + (c >> 2)) * K + kt + ((c & 3) << 3));
    }
    if (wave == 0)
      tdm_load_tile(Bw, bn0, kt, BN2, N, K, BK, asBo[buf]);
  };

  stage(0, 0);
  int buf = 0;
  for (int kt = 0; kt < K; kt += BK) {
    const bool has_next = (kt + BK) < K;
    if (has_next) {
      stage(buf ^ 1, kt + BK);              // overlap next slab with compute
      if (kt + 2 * BK < K) {                // warm caches for the slab after
        __builtin_prefetch(A  + (bm0 + (tid >> 2)) * K + kt + 2 * BK, 0, 3);
        __builtin_prefetch(Bw + (bn0 + (tid >> 2)) * K + kt + 2 * BK, 0, 3);
      }
      wait_async2();                        // drain stage i (in-order), keep i+1
      if (wave == 0) __builtin_amdgcn_s_wait_tensorcnt((short)1);
    } else {
      wait_async0();
      if (wave == 0) __builtin_amdgcn_s_wait_tensorcnt((short)0);
    }
    __syncthreads();

    // A fragments: lane ln = row, 16 contiguous K elements [hh*16..]
    v16bf af[4];
#pragma unroll
    for (int mi = 0; mi < 4; ++mi) {
      const int row = wm * 64 + mi * 16 + ln;
      const int lo  = hh * 16;
      v8bf a0 = *(const v8bf*)&As[buf][row * BK + lo];
      v8bf a1 = *(const v8bf*)&As[buf][row * BK + 8 + lo];
#pragma unroll
      for (int e = 0; e < 8; ++e) { af[mi][e] = a0[e]; af[mi][8 + e] = a1[e]; }
    }
    // B fragments: lane ln = output column, 16 contiguous K elements
    v16bf bfr[4];
#pragma unroll
    for (int ni = 0; ni < 4; ++ni) {
      const int row = wn * 64 + ni * 16 + ln;
      bfr[ni] = *(const v16bf*)&Bs[buf][row * BK + hh * 16];
    }
#pragma unroll
    for (int mi = 0; mi < 4; ++mi)
#pragma unroll
      for (int ni = 0; ni < 4; ++ni)
        acc[mi][ni] = wmma_bf16(af[mi], bfr[ni], acc[mi][ni]);
    __syncthreads();
    buf ^= 1;
  }

  // C layout: vgpr r, lane -> element (r + 8*hh, ln)
#pragma unroll
  for (int mi = 0; mi < 4; ++mi)
#pragma unroll
    for (int ni = 0; ni < 4; ++ni) {
      const long m0 = bm0 + wm * 64 + mi * 16 + hh * 8;
      const long n0 = bn0 + wn * 64 + ni * 16 + ln;
#pragma unroll
      for (int r = 0; r < 8; ++r)
        C[(m0 + r) * N + n0] = acc[mi][ni][r];
    }
}

// ============================================================================
// Flash attention (causal, GQA). Q (B,H,S,D) bf16, K (B,KV,S,D) bf16,
// Vt (B,KV,D,S) bf16  ->  O (B,S,H*D) bf16.  One wave = 16 query rows;
// block = 8 waves = 128 query rows; key blocks of 32; 16 WMMAs / iteration.
// ============================================================================
__global__ __launch_bounds__(256) void flash_attn(
    const __bf16* __restrict__ Q, const __bf16* __restrict__ Kc,
    const __bf16* __restrict__ Vt, __bf16* __restrict__ O)
{
  __shared__ __align__(64) __bf16 pbuf[8][16 * 32];  // per-wave P scratch
  const int lane = threadIdx.x & 31;
  const int wave = threadIdx.x >> 5;
  const int ln   = lane & 15;
  const int hh   = lane >> 4;
  const int b    = blockIdx.z;
  const int h    = blockIdx.y;
  const int kvh  = h / GQA;
  const int q0   = blockIdx.x * 128 + wave * 16;

  const __bf16* Qb = Q  + ((size_t)(b * NHEADS + h) * SEQ + q0) * HD;
  const __bf16* Kb = Kc + (size_t)(b * NKV + kvh) * SEQ * HD;
  const __bf16* Vb = Vt + (size_t)(b * NKV + kvh) * HD * SEQ;

  // Q A-fragments over the 128-wide head dim: 4 chunks of K=32
  v16bf qf[4];
#pragma unroll
  for (int dc = 0; dc < 4; ++dc) {
    const int lo = dc * 32 + hh * 16;
    v8bf x0 = *(const v8bf*)(Qb + (size_t)ln * HD + lo);
    v8bf x1 = *(const v8bf*)(Qb + (size_t)ln * HD + lo + 8);
#pragma unroll
    for (int e = 0; e < 8; ++e) { qf[dc][e] = x0[e]; qf[dc][8 + e] = x1[e]; }
  }

  v8f o[8];
#pragma unroll
  for (int t = 0; t < 8; ++t) o[t] = (v8f)(0.0f);
  float mrow[8], lrow[8];
#pragma unroll
  for (int r = 0; r < 8; ++r) { mrow[r] = -1e30f; lrow[r] = 0.0f; }

  const float scale = 0.08838834764831845f;  // 1/sqrt(128)
  const int   kend  = q0 + 16;               // causal: keys <= q0+15 needed

  for (int kb = 0; kb < kend; kb += 32) {
    // ---- scores: two 16x16 tiles over 32 keys, contraction over D=128
    v8f s0 = (v8f)(0.0f), s1 = (v8f)(0.0f);
#pragma unroll
    for (int dc = 0; dc < 4; ++dc) {
      v16bf k0 = *(const v16bf*)(Kb + (size_t)(kb + ln) * HD + dc * 32 + hh * 16);
      s0 = wmma_bf16(qf[dc], k0, s0);
    }
#pragma unroll
    for (int dc = 0; dc < 4; ++dc) {
      v16bf k1 = *(const v16bf*)(Kb + (size_t)(kb + 16 + ln) * HD + dc * 32 + hh * 16);
      s1 = wmma_bf16(qf[dc], k1, s1);
    }

    // ---- online softmax across the 16-lane column groups
    const bool needmask = (kb + 31 > q0);
    float corr[8];
#pragma unroll
    for (int r = 0; r < 8; ++r) {
      const int qrow = q0 + r + 8 * hh;
      float a  = s0[r] * scale;
      float c2 = s1[r] * scale;
      if (needmask) {
        if (kb + ln      > qrow) a  = -1e30f;
        if (kb + 16 + ln > qrow) c2 = -1e30f;
      }
      const float mx   = redmax16(fmaxf(a, c2));
      const float mnew = fmaxf(mrow[r], mx);
      corr[r] = __expf(mrow[r] - mnew);
      mrow[r] = mnew;
      a  = __expf(a  - mnew);
      c2 = __expf(c2 - mnew);
      lrow[r] = lrow[r] * corr[r] + redsum16(a + c2);
      s0[r] = a; s1[r] = c2;
    }
#pragma unroll
    for (int t = 0; t < 8; ++t)
#pragma unroll
      for (int r = 0; r < 8; ++r) o[t][r] *= corr[r];

    // ---- C-layout -> A-layout via per-wave LDS round trip (in-order DS)
    __bf16* pw = &pbuf[wave][0];
#pragma unroll
    for (int r = 0; r < 8; ++r) {
      pw[(r + 8 * hh) * 32 + ln]      = (__bf16)s0[r];
      pw[(r + 8 * hh) * 32 + 16 + ln] = (__bf16)s1[r];
    }
    v16bf pf;
    {
      const int lo = hh * 16;
      v8bf p0 = *(const v8bf*)&pw[ln * 32 + lo];
      v8bf p1 = *(const v8bf*)&pw[ln * 32 + 8 + lo];
#pragma unroll
      for (int e = 0; e < 8; ++e) { pf[e] = p0[e]; pf[8 + e] = p1[e]; }
    }

    // ---- O += P(16x32) * V(32x16 per 16-col d-tile), V stored transposed
#pragma unroll
    for (int t = 0; t < 8; ++t) {
      v16bf vf = *(const v16bf*)(Vb + (size_t)(t * 16 + ln) * SEQ + kb + hh * 16);
      o[t] = wmma_bf16(pf, vf, o[t]);
    }
  }

  float inv[8];
#pragma unroll
  for (int r = 0; r < 8; ++r) inv[r] = 1.0f / lrow[r];
#pragma unroll
  for (int t = 0; t < 8; ++t)
#pragma unroll
    for (int r = 0; r < 8; ++r) {
      const int m = q0 + r + 8 * hh;
      O[(size_t)(b * SEQ + m) * HIDDEN + h * HD + t * 16 + ln] =
          (__bf16)(o[t][r] * inv[r]);
    }
}

// ============================================================================
// RoPE: read f32 projection (BS, heads*128), rotate, write bf16 (B,heads,S,128)
// ============================================================================
__global__ void rope_pack(const float* __restrict__ proj, const int* __restrict__ pos,
                          __bf16* __restrict__ out, int heads)
{
  const long idx   = (long)blockIdx.x * blockDim.x + threadIdx.x;
  const long total = (long)BS * heads * 64;
  if (idx >= total) return;
  const int  i = (int)(idx & 63);
  const int  h = (int)((idx >> 6) % heads);
  const long t = idx / (64 * heads);               // t = b*SEQ + s
  const int  p = pos[t];
  // inv_freq = theta^(-i/64) = exp(-ln(theta) * i / 64)
  const float freq = (float)p * __expf(-0.015625f * (float)i * __logf(10000.0f));
  const float cs = __cosf(freq), sn = __sinf(freq);
  const float x1 = proj[t * ((long)heads * HD) + (long)h * HD + i];
  const float x2 = proj[t * ((long)heads * HD) + (long)h * HD + i + 64];
  const long bb = t / SEQ, s = t % SEQ;
  const long ob = ((bb * heads + h) * SEQ + s) * HD;
  out[ob + i]      = (__bf16)(x1 * cs - x2 * sn);
  out[ob + 64 + i] = (__bf16)(x2 * cs + x1 * sn);
}

// V: f32 (BS, KVDIM) -> bf16 transposed (B, KV, D, S)
__global__ void v_transpose(const float* __restrict__ proj, __bf16* __restrict__ vt)
{
  const long idx   = (long)blockIdx.x * blockDim.x + threadIdx.x;
  const long total = (long)BATCH * NKV * HD * SEQ;
  if (idx >= total) return;
  const long s  = idx % SEQ;
  const long d  = (idx / SEQ) % HD;
  const long kv = (idx / ((long)SEQ * HD)) % NKV;
  const long b  = idx / ((long)SEQ * HD * NKV);
  vt[idx] = (__bf16)proj[(b * SEQ + s) * KVDIM + kv * HD + d];
}

__global__ void cvt_bf16(const float* __restrict__ in, __bf16* __restrict__ out, long n)
{
  const long i = (long)blockIdx.x * blockDim.x + threadIdx.x;
  if (i < n) out[i] = (__bf16)in[i];
}

// ============================================================================
extern "C" void kernel_launch(void* const* d_in, const int* in_sizes, int n_in,
                              void* d_out, int out_size, void* d_ws, size_t ws_size,
                              hipStream_t stream)
{
  const float* X  = (const float*)d_in[0];
  const int*   P  = (const int*)  d_in[1];
  const float* Wq = (const float*)d_in[2];
  const float* Wk = (const float*)d_in[3];
  const float* Wv = (const float*)d_in[4];
  const float* Wo = (const float*)d_in[5];
  float* out = (float*)d_out;

  char* ws = (char*)d_ws;
  size_t off = 0;
  auto alloc = [&](size_t bytes) -> char* {
    char* p = ws + off;
    off += (bytes + 255) & ~(size_t)255;
    return p;
  };
  __bf16* Xb  = (__bf16*)alloc((size_t)BS * HIDDEN * 2);
  __bf16* Wqb = (__bf16*)alloc((size_t)HIDDEN * HIDDEN * 2);
  __bf16* Wkb = (__bf16*)alloc((size_t)KVDIM * HIDDEN * 2);
  __bf16* Wvb = (__bf16*)alloc((size_t)KVDIM * HIDDEN * 2);
  __bf16* Wob = (__bf16*)alloc((size_t)HIDDEN * HIDDEN * 2);
  float*  Qp  = (float*) alloc((size_t)BS * HIDDEN * 4);
  float*  Kp  = (float*) alloc((size_t)BS * KVDIM * 4);
  float*  Vp  = (float*) alloc((size_t)BS * KVDIM * 4);
  __bf16* Qr  = (__bf16*)alloc((size_t)BATCH * NHEADS * SEQ * HD * 2);
  __bf16* Kr  = (__bf16*)alloc((size_t)BATCH * NKV * SEQ * HD * 2);
  __bf16* Vtb = (__bf16*)alloc((size_t)BATCH * NKV * HD * SEQ * 2);
  __bf16* Ob  = (__bf16*)alloc((size_t)BS * HIDDEN * 2);

  auto cvt = [&](const float* src, __bf16* dst, long n) {
    cvt_bf16<<<dim3((unsigned)((n + 255) / 256)), dim3(256), 0, stream>>>(src, dst, n);
  };
  cvt(X,  Xb,  (long)BS * HIDDEN);
  cvt(Wq, Wqb, (long)HIDDEN * HIDDEN);
  cvt(Wk, Wkb, (long)KVDIM * HIDDEN);
  cvt(Wv, Wvb, (long)KVDIM * HIDDEN);
  cvt(Wo, Wob, (long)HIDDEN * HIDDEN);

  // QKV projections (all dims are multiples of the 128x256 block tile)
  gemm_abt_bf16<<<dim3(HIDDEN / BN2, BS / BM), dim3(256), 0, stream>>>(Xb, Wqb, Qp, BS, HIDDEN, HIDDEN);
  gemm_abt_bf16<<<dim3(KVDIM  / BN2, BS / BM), dim3(256), 0, stream>>>(Xb, Wkb, Kp, BS, KVDIM,  HIDDEN);
  gemm_abt_bf16<<<dim3(KVDIM  / BN2, BS / BM), dim3(256), 0, stream>>>(Xb, Wvb, Vp, BS, KVDIM,  HIDDEN);

  rope_pack<<<dim3((unsigned)(((long)BS * NHEADS * 64 + 255) / 256)), dim3(256), 0, stream>>>(Qp, P, Qr, NHEADS);
  rope_pack<<<dim3((unsigned)(((long)BS * NKV    * 64 + 255) / 256)), dim3(256), 0, stream>>>(Kp, P, Kr, NKV);
  v_transpose<<<dim3((unsigned)(((long)BATCH * NKV * HD * SEQ + 255) / 256)), dim3(256), 0, stream>>>(Vp, Vtb);

  flash_attn<<<dim3(SEQ / 128, NHEADS, BATCH), dim3(256), 0, stream>>>(Qr, Kr, Vtb, Ob);

  // output projection straight into d_out (f32)
  gemm_abt_bf16<<<dim3(HIDDEN / BN2, BS / BM), dim3(256), 0, stream>>>(Ob, Wob, out, BS, HIDDEN, HIDDEN);

  (void)in_sizes; (void)n_in; (void)out_size; (void)ws_size;
}
